// TorchModel_68410239090831
// MI455X (gfx1250) — compile-verified, matrix operand order checked by reading, not run on hardware
//
#include <hip/hip_runtime.h>
#include <hip/hip_bf16.h>
#include <stdint.h>

typedef __attribute__((ext_vector_type(16))) _Float16 v16h;
typedef __attribute__((ext_vector_type(8)))  _Float16 v8h;
typedef __attribute__((ext_vector_type(8)))  float    v8f;
typedef __attribute__((ext_vector_type(4)))  float    f32x4;

static constexpr int H   = 512;
static constexpr int C   = 9;
static constexpr int Bc  = 128;
static constexpr int Tc  = 256;
static constexpr int G4H = 2048;   // 4*H gate width

__device__ __forceinline__ v16h cat16(v8h lo, v8h hi) {
  return __builtin_shufflevector(lo, hi, 0,1,2,3,4,5,6,7,8,9,10,11,12,13,14,15);
}

// ---------------------------------------------------------------------------
// Embedding gather + max-norm clip -> xs[t*B+b][512] (f16)
// ---------------------------------------------------------------------------
__global__ __launch_bounds__(128) void k_embed(const int* __restrict__ x,
                                               const float* __restrict__ embed,
                                               _Float16* __restrict__ xs) {
  const int rid = blockIdx.x;        // rid = t*B + b
  const int t   = rid >> 7;
  const int b   = rid & 127;
  const int tid = threadIdx.x;
  const int tok = x[b * Tc + t];
  f32x4 v = *(const f32x4*)(embed + (size_t)tok * H + tid * 4);
  float ss = v.x*v.x + v.y*v.y + v.z*v.z + v.w*v.w;
  #pragma unroll
  for (int off = 16; off > 0; off >>= 1) ss += __shfl_xor(ss, off, 32);
  __shared__ float sred[4];
  if ((tid & 31) == 0) sred[tid >> 5] = ss;
  __syncthreads();
  const float nrm = sqrtf(sred[0] + sred[1] + sred[2] + sred[3]);
  const float sc  = (nrm > 1.f) ? (1.f / (nrm + 1e-7f)) : 1.f;
  _Float16* dst = xs + (size_t)rid * H + tid * 4;
  dst[0] = (_Float16)(v.x * sc); dst[1] = (_Float16)(v.y * sc);
  dst[2] = (_Float16)(v.z * sc); dst[3] = (_Float16)(v.w * sc);
}

// ---------------------------------------------------------------------------
// Pack [W_ih ; W_hh] (f32, row = output gate n, col = k) into WMMA B-fragment
// order: frag(kt,nt) = 32 lanes x 16 halves, contiguous.  out[idx] layout ==
// ((kt*128 + nt)*32 + lane)*16 + e.
// ---------------------------------------------------------------------------
__global__ __launch_bounds__(256) void k_pack(const float* __restrict__ w_ih,
                                              const float* __restrict__ w_hh,
                                              _Float16* __restrict__ wp,
                                              const int KIN) {
  const int idx  = blockIdx.x * 256 + threadIdx.x;
  const int e    = idx & 15;
  const int lane = (idx >> 4) & 31;
  const int nt   = (idx >> 9) & 127;
  const int kt   = idx >> 16;
  const int kb   = (lane & 16) ? 8 : 0;
  const int k    = kt * 32 + kb + (e < 8 ? e : e + 8);
  const int n    = nt * 16 + (lane & 15);
  const float v  = (k < KIN) ? w_ih[(size_t)n * KIN + k]
                             : w_hh[(size_t)n * H + (k - KIN)];
  wp[idx] = (_Float16)v;
}

// ---------------------------------------------------------------------------
// Fused LSTM scan: one block = 16 batch rows, all T steps.
//   z = [x_t | h] * Wcat^T + b   (WMMA f16->f32, KCAT = KIN + 512)
// Wave w owns N-tiles {w, w+16, ..., w+112}; tiles (ti), (ti+2), (ti+4),
// (ti+6) are the i/f/g/o gates of the SAME (b,j) in the same lane slot, so
// the cell update is pure register math.  blockIdx.y = direction.
// ---------------------------------------------------------------------------
template <int KIN>
__global__ __launch_bounds__(512) void k_lstm_scan(
    const _Float16* __restrict__ xin,   // [T][B][KIN] f16
    const _Float16* __restrict__ wpall, // packed, per-dir stride G4H*KCAT
    const float* __restrict__ ball,     // [2][2048]
    _Float16* __restrict__ hout) {      // [T][B][1024] f16
  constexpr int KCAT = KIN + H;
  constexpr int KT   = KCAT / 32;
  __shared__ __attribute__((aligned(16))) _Float16 xh[16][KCAT];

  const int tid    = threadIdx.x;
  const int wave   = tid >> 5;
  const int lane   = tid & 31;
  const int lane15 = lane & 15;
  const int kb     = (lane & 16) ? 8 : 0;
  const int batch0 = blockIdx.x * 16;
  const int dir    = blockIdx.y;
  const _Float16* wp   = wpall + (size_t)dir * (size_t)G4H * KCAT;
  const float*    bias = ball + dir * G4H;
  const int       hoff = dir * H;

  // zero the h region of the LDS tile
  const v8h hz = {0, 0, 0, 0, 0, 0, 0, 0};
  for (int q = tid; q < 16 * H / 8; q += 512) {
    const int row = q >> 6;
    const int col = (q & 63) * 8;
    *(v8h*)&xh[row][KIN + col] = hz;
  }
  float creg[16];
  #pragma unroll
  for (int i = 0; i < 16; ++i) creg[i] = 0.f;
  float bs[8];
  #pragma unroll
  for (int i = 0; i < 8; ++i) bs[i] = bias[(wave + 16 * i) * 16 + lane15];
  __syncthreads();

  for (int s = 0; s < Tc; ++s) {
    const int t = dir ? (Tc - 1 - s) : s;
    // --- async-stage x_t tile into LDS (CDNA5 ASYNCcnt path) ---
    const _Float16* xsrc = xin + ((size_t)t * Bc + batch0) * KIN;
    constexpr int CH  = 16 * KIN / 8;  // 16-byte chunks
    constexpr int CPR = KIN / 8;       // chunks per row
    for (int q = tid; q < CH; q += 512) {
      const int row = q / CPR;
      const int col = (q % CPR) * 8;
      unsigned laddr = (unsigned)(size_t)&xh[row][col];
      unsigned long long gaddr =
          (unsigned long long)(size_t)(xsrc + (size_t)row * KIN + col);
      asm volatile("global_load_async_to_lds_b128 %0, %1, off"
                   :: "v"(laddr), "v"(gaddr) : "memory");
    }
    asm volatile("s_wait_asynccnt 0" ::: "memory");
    __syncthreads();

    // --- WMMA: z = [x|h] * Wcat^T ---
    v8f acc[8];
    const v8f vz = {0.f, 0.f, 0.f, 0.f, 0.f, 0.f, 0.f, 0.f};
    #pragma unroll
    for (int i = 0; i < 8; ++i) acc[i] = vz;

    for (int kt = 0; kt < KT; ++kt) {
      const _Float16* ap = &xh[lane15][kt * 32 + kb];
      const v16h a = cat16(*(const v8h*)ap, *(const v8h*)(ap + 16));
      const _Float16* bp = wp + (((size_t)kt * 128 + wave) * 32 + lane) * 16;
      #pragma unroll
      for (int i = 0; i < 8; ++i) {
        const _Float16* bpi = bp + (size_t)i * 8192;  // +16 N-tiles
        const v16h bm = cat16(*(const v8h*)bpi, *(const v8h*)(bpi + 8));
        acc[i] = __builtin_amdgcn_wmma_f32_16x16x32_f16(
            false, a, false, bm, (short)0, acc[i], false, false);
      }
    }
    __syncthreads();  // all waves done reading xh

    // --- gate math in registers, write h to LDS + global ---
    const int mbase = (lane >> 4) * 8;
    _Float16* hdst = hout + ((size_t)t * Bc + batch0) * (2 * H) + hoff;
    #pragma unroll
    for (int ti = 0; ti < 2; ++ti) {
      const int j = (wave + 16 * ti) * 16 + lane15;   // hidden index 0..511
      #pragma unroll
      for (int r = 0; r < 8; ++r) {
        const float zi = acc[ti    ][r] + bs[ti    ];
        const float zf = acc[ti + 2][r] + bs[ti + 2];
        const float zg = acc[ti + 4][r] + bs[ti + 4];
        const float zo = acc[ti + 6][r] + bs[ti + 6];
        float c = creg[ti * 8 + r];
        const float ig = 1.f / (1.f + expf(-zi));
        const float fg = 1.f / (1.f + expf(-zf));
        const float og = 1.f / (1.f + expf(-zo));
        c = fg * c + ig * tanhf(zg);
        const float hval = og * tanhf(c);
        creg[ti * 8 + r] = c;
        const _Float16 hh = (_Float16)hval;
        const int m = mbase + r;
        xh[m][KIN + j] = hh;
        hdst[(size_t)m * (2 * H) + j] = hh;
      }
    }
    __syncthreads();
  }
}

// ---------------------------------------------------------------------------
// Emissions: 9-class classifier over 1024-dim hs (too small-N for WMMA)
// ---------------------------------------------------------------------------
__global__ __launch_bounds__(256) void k_emissions(
    const _Float16* __restrict__ hs, const float* __restrict__ cls_w,
    const float* __restrict__ cls_b, float* __restrict__ em) {
  const int lane = threadIdx.x & 31;
  const int row  = blockIdx.x * 8 + (threadIdx.x >> 5);
  const _Float16* hp = hs + (size_t)row * (2 * H);
  float s[C];
  #pragma unroll
  for (int c = 0; c < C; ++c) s[c] = 0.f;
  for (int k = lane; k < 2 * H; k += 32) {
    const float hv = (float)hp[k];
    #pragma unroll
    for (int c = 0; c < C; ++c) s[c] += hv * cls_w[c * (2 * H) + k];
  }
  #pragma unroll
  for (int c = 0; c < C; ++c) {
    #pragma unroll
    for (int off = 16; off > 0; off >>= 1) s[c] += __shfl_xor(s[c], off, 32);
  }
  if (lane == 0) {
    #pragma unroll
    for (int c = 0; c < C; ++c) em[(size_t)row * C + c] = s[c] + cls_b[c];
  }
}

// ---------------------------------------------------------------------------
// CRF numerator + forward-algorithm denominator + mean NLL (one block, b/thread)
// ---------------------------------------------------------------------------
__global__ __launch_bounds__(128) void k_crf(const float* __restrict__ em,
                                             const int* __restrict__ y,
                                             const float* __restrict__ cstart,
                                             const float* __restrict__ cend,
                                             const float* __restrict__ ctrans,
                                             float* __restrict__ out) {
  const int b = threadIdx.x;
  __shared__ float sT[C * C], sS[C], sE[C], red[Bc];
  if (b < C * C) sT[b] = ctrans[b];
  if (b < C) { sS[b] = cstart[b]; sE[b] = cend[b]; }
  __syncthreads();

  const int y0 = y[b * Tc];
  const int tag0 = y0 < 0 ? 0 : y0;
  float alpha[C], nxt[C];
  #pragma unroll
  for (int c = 0; c < C; ++c) alpha[c] = sS[c] + em[b * C + c];
  float score = sS[tag0] + em[b * C + tag0];
  int prev = tag0;
  int cnt = (y0 > -1) ? 1 : 0;

  for (int t = 1; t < Tc; ++t) {
    const int yv = y[b * Tc + t];
    const bool mt = yv > -1;
    const int tg = yv < 0 ? 0 : yv;
    const float* et = em + ((size_t)t * Bc + b) * C;
    if (mt) {
      score += sT[prev * C + tg] + et[tg];
      ++cnt;
      #pragma unroll
      for (int j = 0; j < C; ++j) {
        float mx = -1e30f;
        #pragma unroll
        for (int i = 0; i < C; ++i) mx = fmaxf(mx, alpha[i] + sT[i * C + j]);
        float sum = 0.f;
        #pragma unroll
        for (int i = 0; i < C; ++i) sum += expf(alpha[i] + sT[i * C + j] - mx);
        nxt[j] = mx + logf(sum) + et[j];
      }
      #pragma unroll
      for (int j = 0; j < C; ++j) alpha[j] = nxt[j];
    }
    prev = tg;
  }
  int le = cnt - 1; if (le < 0) le = 0;
  int lastv = y[b * Tc + le]; if (lastv < 0) lastv = 0;
  score += sE[lastv];
  float mx = -1e30f;
  #pragma unroll
  for (int c = 0; c < C; ++c) mx = fmaxf(mx, alpha[c] + sE[c]);
  float sum = 0.f;
  #pragma unroll
  for (int c = 0; c < C; ++c) sum += expf(alpha[c] + sE[c] - mx);
  red[b] = score - (mx + logf(sum));
  __syncthreads();
  if (b == 0) {
    float acc = 0.f;
    for (int i = 0; i < Bc; ++i) acc += red[i];
    out[0] = -acc / (float)Bc;
  }
}

// ---------------------------------------------------------------------------
extern "C" void kernel_launch(void* const* d_in, const int* in_sizes, int n_in,
                              void* d_out, int out_size, void* d_ws,
                              size_t ws_size, hipStream_t stream) {
  (void)in_sizes; (void)n_in; (void)out_size; (void)ws_size;
  const int*   x       = (const int*)  d_in[0];
  const int*   y       = (const int*)  d_in[1];
  const float* embed   = (const float*)d_in[2];
  const float* w_ih_l0 = (const float*)d_in[3];
  const float* w_hh_l0 = (const float*)d_in[4];
  const float* b_l0    = (const float*)d_in[5];
  const float* w_ih_l1 = (const float*)d_in[6];
  const float* w_hh_l1 = (const float*)d_in[7];
  const float* b_l1    = (const float*)d_in[8];
  const float* cls_w   = (const float*)d_in[9];
  const float* cls_b   = (const float*)d_in[10];
  const float* crf_s   = (const float*)d_in[11];
  const float* crf_e   = (const float*)d_in[12];
  const float* crf_t   = (const float*)d_in[13];
  float* out = (float*)d_out;

  char* ws = (char*)d_ws;
  size_t off = 0;
  auto take = [&](size_t bytes) -> char* {
    char* p = ws + off;
    off += (bytes + 255) & ~(size_t)255;
    return p;
  };
  _Float16* xs0   = (_Float16*)take((size_t)Tc * Bc * H * 2);
  _Float16* hs0   = (_Float16*)take((size_t)Tc * Bc * 2 * H * 2);
  _Float16* hs1   = (_Float16*)take((size_t)Tc * Bc * 2 * H * 2);
  _Float16* wp_l0 = (_Float16*)take((size_t)2 * G4H * 1024 * 2);
  _Float16* wp_l1 = (_Float16*)take((size_t)2 * G4H * 1536 * 2);
  float*    em    = (float*)   take((size_t)Tc * Bc * C * 4);

  k_embed<<<Tc * Bc, 128, 0, stream>>>(x, embed, xs0);
  k_pack<<<(G4H * 1024) / 256, 256, 0, stream>>>(w_ih_l0, w_hh_l0, wp_l0, H);
  k_pack<<<(G4H * 1024) / 256, 256, 0, stream>>>(
      w_ih_l0 + (size_t)G4H * H, w_hh_l0 + (size_t)G4H * H,
      wp_l0 + (size_t)G4H * 1024, H);
  k_pack<<<(G4H * 1536) / 256, 256, 0, stream>>>(w_ih_l1, w_hh_l1, wp_l1, 2 * H);
  k_pack<<<(G4H * 1536) / 256, 256, 0, stream>>>(
      w_ih_l1 + (size_t)G4H * 2 * H, w_hh_l1 + (size_t)G4H * H,
      wp_l1 + (size_t)G4H * 1536, 2 * H);

  k_lstm_scan<512><<<dim3(8, 2), 512, 0, stream>>>(xs0, wp_l0, b_l0, hs0);
  k_lstm_scan<1024><<<dim3(8, 2), 512, 0, stream>>>(hs0, wp_l1, b_l1, hs1);

  k_emissions<<<(Tc * Bc) / 8, 256, 0, stream>>>(hs1, cls_w, cls_b, em);
  k_crf<<<1, Bc, 0, stream>>>(em, y, crf_s, crf_e, crf_t, out);
}
